// ResearchGNN_13623636263493
// MI455X (gfx1250) — matrix operand order repaired and checked by reading, not compiled, not run on hardware
//
#include <hip/hip_runtime.h>
#include <hip/hip_bf16.h>

typedef __attribute__((ext_vector_type(16))) __bf16 v16bf;
typedef __attribute__((ext_vector_type(8)))  float  v8f;

#define HID   128
#define NCLS  5
#define INDIM 5
#define NGR   64

// ---------------- weight conversion ----------------
__global__ void k_convW(const float* __restrict__ Wg, __bf16* __restrict__ WgB, int n) {
    int i = blockIdx.x * blockDim.x + threadIdx.x;
    if (i < n) WgB[i] = (__bf16)Wg[i];
}

// ---------------- degree / norm ----------------
__global__ void k_deg_init(float* deg, int N) {
    int i = blockIdx.x * blockDim.x + threadIdx.x;
    if (i < N) deg[i] = 1.0f; // self loop
}
__global__ void k_deg_accum(const int* __restrict__ ei, float* deg, int E) {
    int e = blockIdx.x * blockDim.x + threadIdx.x;
    if (e < E) atomicAdd(&deg[ei[E + e]], 1.0f); // dst row
}
__global__ void k_to_dinv(float* deg, int N) {
    int i = blockIdx.x * blockDim.x + threadIdx.x;
    if (i < N) deg[i] = rsqrtf(deg[i]);
}

// ---------------- input projection: h = x @ Wp + bp ----------------
__global__ void k_proj(const float* __restrict__ x, const float* __restrict__ Wp,
                       const float* __restrict__ bp, float* __restrict__ h,
                       __bf16* __restrict__ hb, int N) {
    int n = blockIdx.x, c = threadIdx.x;
    __shared__ float xs[INDIM];
    if (c < INDIM) xs[c] = x[n * INDIM + c];
    __syncthreads();
    float acc = bp[c];
    #pragma unroll
    for (int k = 0; k < INDIM; ++k) acc += xs[k] * Wp[k * HID + c];
    h[n * HID + c]  = acc;
    hb[n * HID + c] = (__bf16)acc;
}

// ---------------- WMMA GEMM: m[N,128] = hb[N,128](bf16) @ Wb[128,128](bf16) ----------------
__global__ void k_gemm_wmma(const __bf16* __restrict__ hb, const __bf16* __restrict__ Wb,
                            float* __restrict__ m, int N) {
    __shared__ __bf16 As[64 * HID];    // A block, row-major [r][k]
    __shared__ __bf16 Bs[HID * HID];   // B transposed: Bs[n*128 + k]
    const int tid = threadIdx.x;
    const int blockRow = blockIdx.x * 64;

    // --- stage A rows via async global->LDS copy engine (16B chunks, ASYNCcnt) ---
    // 64*128 bf16 = 1024 chunks of 8 bf16; 4 chunks per thread.
    for (int idx = tid; idx < (64 * HID) / 8; idx += 256) {
        int r = idx >> 4;                      // 16 chunks per row
        int gr = blockRow + r;
        if (gr >= N) gr = N - 1;               // clamp: row-local data, masked at store
        const __bf16* gsrc = hb + (size_t)gr * HID + (idx & 15) * 8;
        unsigned lds = (unsigned)(size_t)(&As[idx * 8]);
        asm volatile("global_load_async_to_lds_b128 %0, %1, off"
                     :: "v"(lds), "v"(gsrc) : "memory");
    }

    // --- stage B with transpose (must pass through VGPRs), overlaps async A copies ---
    for (int idx = tid; idx < HID * HID; idx += 256) {
        int k = idx >> 7, n = idx & 127;
        Bs[n * HID + k] = Wb[idx];
    }

    asm volatile("s_wait_asynccnt 0x0" ::: "memory");
    __syncthreads();

    const int lane = tid & 31;
    const int wave = tid >> 5;
    const int rowTile = (wave >> 1) * 16;      // 0,16,32,48
    const int colBase = (wave & 1) * 64;       // 0 or 64
    const int r    = lane & 15;
    const int half = lane >> 4;

    v8f acc[4] = {};

    union FragU { v16bf v; uint4 q[2]; };

    for (int kk = 0; kk < HID; kk += 32) {
        // A fragment (16x32 bf16): lane holds row M=r;
        // run0: K = kk + half*8 + [0..7], run1: K = kk + 16 + half*8 + [0..7]
        FragU A;
        const __bf16* arow = &As[(rowTile + r) * HID + kk + half * 8];
        A.q[0] = *(const uint4*)(arow);
        A.q[1] = *(const uint4*)(arow + 16);
        #pragma unroll
        for (int t = 0; t < 4; ++t) {
            // B fragment (32x16 bf16): lane holds col N=r; run0: K = kk + half*8 + [0..7], run1: K = kk + 16 + half*8 + [0..7]
            FragU B;
            const __bf16* bcol = &Bs[(colBase + t * 16 + r) * HID + kk + half * 8];
            B.q[0] = *(const uint4*)(bcol);
            B.q[1] = *(const uint4*)(bcol + 16);
            acc[t] = __builtin_amdgcn_wmma_f32_16x16x32_bf16(
                false, A.v, false, B.v, (short)0, acc[t], false, false);
        }
    }

    // C/D layout: VGPR v: M = v + half*8 ; N = lane&15
    #pragma unroll
    for (int t = 0; t < 4; ++t) {
        #pragma unroll
        for (int v = 0; v < 8; ++v) {
            int grow = blockRow + rowTile + half * 8 + v;
            if (grow < N) m[grow * HID + colBase + t * 16 + r] = acc[t][v];
        }
    }
}

// ---------------- aggregation ----------------
__global__ void k_agg_init(float* __restrict__ agg, const float* __restrict__ bg, int N) {
    int n = blockIdx.x, c = threadIdx.x;
    agg[n * HID + c] = bg[c];
}
__global__ void k_scatter(const int* __restrict__ ei, const float* __restrict__ dinv,
                          const float* __restrict__ m, float* __restrict__ agg,
                          int E, int N) {
    int e = blockIdx.x, c = threadIdx.x;
    int s, d; float w;
    if (e < E) { s = ei[e]; d = ei[E + e]; w = dinv[s] * dinv[d]; }
    else       { s = d = e - E; float di = dinv[s]; w = di * di; }
    if (c == 0) __builtin_prefetch(m + (size_t)s * HID, 0, 1); // global_prefetch_b8: pull gathered row
    atomicAdd(&agg[d * HID + c], m[s * HID + c] * w);
}

// ---------------- batch norm ----------------
__global__ void k_bn_stats(const float* __restrict__ agg, float* __restrict__ mu,
                           float* __restrict__ rinv, int N) {
    int c = blockIdx.x;       // feature
    int t = threadIdx.x;      // 256 threads
    float s = 0.f, s2 = 0.f;
    for (int n = t; n < N; n += 256) {
        float v = agg[n * HID + c];
        s += v; s2 += v * v;
    }
    __shared__ float sh[256], sh2[256];
    sh[t] = s; sh2[t] = s2;
    __syncthreads();
    for (int off = 128; off > 0; off >>= 1) {
        if (t < off) { sh[t] += sh[t + off]; sh2[t] += sh2[t + off]; }
        __syncthreads();
    }
    if (t == 0) {
        float mean = sh[0] / (float)N;
        float var  = sh2[0] / (float)N - mean * mean;
        mu[c]   = mean;
        rinv[c] = rsqrtf(var + 1e-5f);
    }
}
__global__ void k_bn_apply(const float* __restrict__ agg, const float* __restrict__ mu,
                           const float* __restrict__ rinv, const float* __restrict__ gamma,
                           const float* __restrict__ beta, float* __restrict__ h,
                           __bf16* __restrict__ hb, int N) {
    int n = blockIdx.x, c = threadIdx.x;
    float v = gamma[c] * (agg[n * HID + c] - mu[c]) * rinv[c] + beta[c];
    v = fmaxf(v, 0.0f);
    h[n * HID + c]  = v;
    hb[n * HID + c] = (__bf16)v;
}

// ---------------- pooling ----------------
__global__ void k_pool_init(float* __restrict__ pooled, float* __restrict__ cnt, int npool) {
    int i = blockIdx.x * blockDim.x + threadIdx.x;
    if (i < npool) pooled[i] = 0.0f;
    if (i < NGR)   cnt[i] = 0.0f;
}
__global__ void k_pool_accum(const float* __restrict__ h, const int* __restrict__ batch,
                             float* __restrict__ pooled, float* __restrict__ cnt, int N) {
    int n = blockIdx.x, c = threadIdx.x;
    int g = batch[n];
    float v = h[n * HID + c];
    atomicAdd(&pooled[g * 2 * HID + c], v);
    // values are post-ReLU (>= 0) so int-ordered max is correct; 0-init matches empty-segment semantics
    atomicMax((int*)&pooled[g * 2 * HID + HID + c], __float_as_int(v));
    if (c == 0) atomicAdd(&cnt[g], 1.0f);
}
__global__ void k_pool_final(float* __restrict__ pooled, const float* __restrict__ cnt) {
    int g = blockIdx.x, c = threadIdx.x;
    pooled[g * 2 * HID + c] /= fmaxf(cnt[g], 1.0f);
}

// ---------------- MLP head ----------------
__global__ void k_mlp(const float* __restrict__ pooled,
                      const float* __restrict__ W1, const float* __restrict__ b1,
                      const float* __restrict__ W2, const float* __restrict__ b2,
                      const float* __restrict__ W3, const float* __restrict__ b3,
                      float* __restrict__ out) {
    int g = blockIdx.x, t = threadIdx.x; // 128 threads
    __shared__ float p[2 * HID], h1[HID], h2[HID / 2];
    p[t]       = pooled[g * 2 * HID + t];
    p[t + HID] = pooled[g * 2 * HID + HID + t];
    __syncthreads();
    float a = b1[t];
    for (int k = 0; k < 2 * HID; ++k) a += p[k] * W1[k * HID + t];
    h1[t] = fmaxf(a, 0.0f);
    __syncthreads();
    if (t < HID / 2) {
        float a2 = b2[t];
        for (int k = 0; k < HID; ++k) a2 += h1[k] * W2[k * (HID / 2) + t];
        h2[t] = fmaxf(a2, 0.0f);
    }
    __syncthreads();
    if (t < NCLS) {
        float a3 = b3[t];
        for (int k = 0; k < HID / 2; ++k) a3 += h2[k] * W3[k * NCLS + t];
        out[g * NCLS + t] = a3;
    }
}

// ---------------- launcher ----------------
extern "C" void kernel_launch(void* const* d_in, const int* in_sizes, int n_in,
                              void* d_out, int out_size, void* d_ws, size_t ws_size,
                              hipStream_t stream) {
    const float* x     = (const float*)d_in[0];
    const int*   ei    = (const int*)d_in[1];
    const int*   batch = (const int*)d_in[2];
    const float* Wp    = (const float*)d_in[3];
    const float* bp    = (const float*)d_in[4];
    const float* Wg    = (const float*)d_in[5];
    const float* bg    = (const float*)d_in[6];
    const float* gamma = (const float*)d_in[7];
    const float* beta  = (const float*)d_in[8];
    const float* W1    = (const float*)d_in[9];
    const float* b1    = (const float*)d_in[10];
    const float* W2    = (const float*)d_in[11];
    const float* b2    = (const float*)d_in[12];
    const float* W3    = (const float*)d_in[13];
    const float* b3    = (const float*)d_in[14];
    float* out = (float*)d_out;

    const int N = in_sizes[2];          // nodes (batch has N entries)
    const int E = in_sizes[1] / 2;      // edges

    // workspace carve
    char* p = (char*)d_ws;
    auto carve = [&](size_t bytes) { void* r = (void*)p; p += (bytes + 255) & ~(size_t)255; return r; };
    float*  dinv   = (float*) carve((size_t)N * 4);
    float*  h      = (float*) carve((size_t)N * HID * 4);
    __bf16* hb     = (__bf16*)carve((size_t)N * HID * 2);
    float*  m      = (float*) carve((size_t)N * HID * 4);
    float*  agg    = (float*) carve((size_t)N * HID * 4);
    __bf16* WgB    = (__bf16*)carve((size_t)4 * HID * HID * 2);
    float*  mu     = (float*) carve(HID * 4);
    float*  rinv   = (float*) carve(HID * 4);
    float*  pooled = (float*) carve((size_t)NGR * 2 * HID * 4);
    float*  cnt    = (float*) carve(NGR * 4);

    const int nW = 4 * HID * HID;
    k_convW<<<(nW + 255) / 256, 256, 0, stream>>>(Wg, WgB, nW);

    k_deg_init <<<(N + 255) / 256, 256, 0, stream>>>(dinv, N);
    k_deg_accum<<<(E + 255) / 256, 256, 0, stream>>>(ei, dinv, E);
    k_to_dinv  <<<(N + 255) / 256, 256, 0, stream>>>(dinv, N);

    k_proj<<<N, HID, 0, stream>>>(x, Wp, bp, h, hb, N);

    const int gemmBlocks = (N + 63) / 64;
    for (int i = 0; i < 4; ++i) {
        k_gemm_wmma<<<gemmBlocks, 256, 0, stream>>>(hb, WgB + (size_t)i * HID * HID, m, N);
        k_agg_init <<<N, HID, 0, stream>>>(agg, bg + i * HID, N);
        k_scatter  <<<E + N, HID, 0, stream>>>(ei, dinv, m, agg, E, N);
        k_bn_stats <<<HID, 256, 0, stream>>>(agg, mu, rinv, N);
        k_bn_apply <<<N, HID, 0, stream>>>(agg, mu, rinv, gamma + i * HID, beta + i * HID, h, hb, N);
    }

    const int npool = NGR * 2 * HID;
    k_pool_init <<<(npool + 255) / 256, 256, 0, stream>>>(pooled, cnt, npool);
    k_pool_accum<<<N, HID, 0, stream>>>(h, batch, pooled, cnt, N);
    k_pool_final<<<NGR, HID, 0, stream>>>(pooled, cnt);

    k_mlp<<<NGR, HID, 0, stream>>>(pooled, W1, b1, W2, b2, W3, b3, out);
}
